// equi_conv2_21603685499531
// MI455X (gfx1250) — matrix operand order repaired and checked by reading, not compiled, not run on hardware
//
#include <hip/hip_runtime.h>
#include <cstdint>
#include <cstddef>

typedef __attribute__((ext_vector_type(16))) _Float16 v16h;
typedef __attribute__((ext_vector_type(8)))  float    v8f;
typedef __attribute__((ext_vector_type(4)))  unsigned int v4u;
typedef __attribute__((ext_vector_type(4)))  int v4s;
typedef __attribute__((ext_vector_type(8)))  int v8s;

#define B_   4
#define P_   1024
#define N_   16
#define C1_  64
#define C2_  64
#define K_   13
#define A_   12
#define KDIM (C1_ * K_ * A_)   /* 9984 contraction length (c,k,a) */
#define NDIM (C2_ * A_)        /* 768  output cols (d,r)          */
#define MDIM (B_ * P_)         /* 4096 output rows (b,p)          */

union FragH { v16h h; uint4 q[2]; };

#if __has_builtin(__builtin_amdgcn_tensor_load_to_lds)
#define USE_TDM 1
#else
#define USE_TDM 0
#endif

// ---------------------------------------------------------------------------
// Kernel 1: build effective weights Wp[nd][kc] in f16, N-major rows so the
// GEMM B tiles are contiguous along K.
//   nd = d*12 + r ; kc = c*156 + k*12 + a
//   Wp[nd][kc] = W[d, c, idx_map[ trace_idxv_rot[r,k]*12 + trace_idx_rot[r,a] ]]
// ---------------------------------------------------------------------------
__global__ __launch_bounds__(256)
void build_w_kernel(const float* __restrict__ W,
                    const int*   __restrict__ idx_map,
                    const int*   __restrict__ tr_kv,   // (A,K)
                    const int*   __restrict__ tr_a,    // (A,A)
                    _Float16*    __restrict__ Wp) {
    int g = blockIdx.x * 256 + threadIdx.x;
    if (g >= NDIM * KDIM) return;
    int nd  = g / KDIM;
    int kc  = g - nd * KDIM;
    int d   = nd / A_;
    int r   = nd - d * A_;
    int c   = kc / (K_ * A_);
    int rem = kc - c * (K_ * A_);
    int k   = rem / A_;
    int a   = rem - k * A_;
    int kk  = tr_kv[r * K_ + k];
    int aa  = tr_a [r * A_ + a];
    int w36 = idx_map[kk * A_ + aa];
    float v = W[((size_t)d * C1_ + c) * 36 + w36];
    Wp[(size_t)nd * KDIM + kc] = (_Float16)v;
}

// ---------------------------------------------------------------------------
// Kernel 2: one workgroup per (b,p). Softmax interpolation weights, neighbor
// feature gather into LDS, emit one 9984-wide f16 row of the GEMM A matrix.
// ---------------------------------------------------------------------------
__global__ __launch_bounds__(256)
void build_a_kernel(const int*   __restrict__ nbr,
                    const float* __restrict__ verts,
                    const float* __restrict__ fm,
                    const float* __restrict__ vs,
                    _Float16*    __restrict__ Aw) {
    __shared__ __align__(16) float s_fn[C1_ * N_ * A_];  // [c][n][a]  48 KB
    __shared__ float s_wts[A_ * N_];
    __shared__ float s_dn [N_ * 3];
    __shared__ float s_vsn[A_ * 3];
    __shared__ int   s_ni [N_];

    int t  = threadIdx.x;
    int bp = blockIdx.x;        // b*1024 + p
    int b  = bp >> 10;
    int p  = bp & (P_ - 1);

    if (t < A_) {               // normalize icosahedron axes
        float x = vs[t * 3 + 0], y = vs[t * 3 + 1], z = vs[t * 3 + 2];
        float n = fmaxf(sqrtf(x * x + y * y + z * z), 1e-12f);
        s_vsn[t * 3 + 0] = x / n;
        s_vsn[t * 3 + 1] = y / n;
        s_vsn[t * 3 + 2] = z / n;
    }
    if (t >= 32 && t < 32 + N_) {  // per-neighbor unit direction
        int n   = t - 32;
        int idx = nbr[(size_t)bp * N_ + n];
        s_ni[n] = idx;
        const float* vc = verts + (size_t)bp * 3;
        const float* vn = verts + ((size_t)b * P_ + idx) * 3;
        float dx = vn[0] - vc[0], dy = vn[1] - vc[1], dz = vn[2] - vc[2];
        float l = fmaxf(sqrtf(dx * dx + dy * dy + dz * dz), 1e-12f);
        s_dn[n * 3 + 0] = dx / l;
        s_dn[n * 3 + 1] = dy / l;
        s_dn[n * 3 + 2] = dz / l;
    }
    __syncthreads();

    if (t < A_ * N_) {          // theta[kk][n] = <vs_norm[kk], dn[n]>
        int kk = t >> 4, n = t & 15;
        s_wts[t] = s_vsn[kk * 3 + 0] * s_dn[n * 3 + 0] +
                   s_vsn[kk * 3 + 1] * s_dn[n * 3 + 1] +
                   s_vsn[kk * 3 + 2] * s_dn[n * 3 + 2];
    }
    __syncthreads();

    if (t < A_) {               // softmax over the 16 neighbors
        float m = -1e30f;
        #pragma unroll
        for (int n = 0; n < N_; ++n) m = fmaxf(m, s_wts[t * N_ + n]);
        float e[N_], sum = 0.f;
        #pragma unroll
        for (int n = 0; n < N_; ++n) { e[n] = __expf(s_wts[t * N_ + n] - m); sum += e[n]; }
        float inv = 1.0f / sum;
        #pragma unroll
        for (int n = 0; n < N_; ++n) s_wts[t * N_ + n] = e[n] * inv;
    }

    for (int i = t; i < C1_ * N_; i += 256) {   // gather neighbor features
        int c = i >> 4, n = i & 15;
        const float4* src = (const float4*)(fm + (((size_t)b * C1_ + c) * P_ + s_ni[n]) * A_);
        float4* dst = (float4*)&s_fn[(c * N_ + n) * A_];
        dst[0] = src[0]; dst[1] = src[1]; dst[2] = src[2];
    }
    __syncthreads();

    _Float16* arow = Aw + (size_t)bp * KDIM;
    for (int idx = t; idx < KDIM; idx += 256) {
        int k   = idx / (C1_ * A_);
        int rem = idx - k * (C1_ * A_);
        int c   = rem / A_;
        int a   = rem - c * A_;
        float acc;
        if (k < A_) {
            const float* w = &s_wts[k * N_];
            const float* f = &s_fn[c * N_ * A_ + a];
            acc = 0.f;
            #pragma unroll
            for (int n = 0; n < N_; ++n) acc += w[n] * f[n * A_];
        } else {
            acc = fm[(((size_t)b * C1_ + c) * P_ + p) * A_ + a];
        }
        arow[c * (K_ * A_) + k * A_ + a] = (_Float16)acc;
    }
}

// ---------------------------------------------------------------------------
// CDNA5 data-movement helpers
// ---------------------------------------------------------------------------
__device__ __forceinline__ void async_copy_b128(const _Float16* g, _Float16* l) {
    // per-lane async copy global -> LDS, tracked by ASYNCcnt.
    // low 32 bits of a generic shared pointer == LDS byte offset on AMDGPU.
    unsigned int lo = (unsigned int)(uintptr_t)l;
    asm volatile("global_load_async_to_lds_b128 %0, %1, off"
                 :: "v"(lo), "v"(g) : "memory");
}

__device__ __forceinline__ void wait_tensor1() {
#if __has_builtin(__builtin_amdgcn_s_wait_tensorcnt)
    __builtin_amdgcn_s_wait_tensorcnt(1);
#else
    asm volatile("s_wait_tensorcnt 1" ::: "memory");
#endif
}
__device__ __forceinline__ void wait_tensor0() {
#if __has_builtin(__builtin_amdgcn_s_wait_tensorcnt)
    __builtin_amdgcn_s_wait_tensorcnt(0);
#else
    asm volatile("s_wait_tensorcnt 0" ::: "memory");
#endif
}

__device__ __forceinline__ int rfl(unsigned int x) {
    return __builtin_amdgcn_readfirstlane((int)x);
}

#if USE_TDM
// One TDM descriptor: 2D tile, 2-byte elements, tile 32 (K) x 128 (N) out of a
// tensor with row stride KDIM, packed linearly into LDS (matches sB[n][k]).
__device__ __forceinline__ void tdm_load_tile(const _Float16* gsrc, _Float16* ldst) {
    unsigned long long ga = (unsigned long long)(uintptr_t)gsrc;
    unsigned int la = (unsigned int)(uintptr_t)ldst;
    v4u g0;
    g0[0] = (unsigned)rfl(1u);                                    // count=1, user D#
    g0[1] = (unsigned)rfl(la);                                    // lds_addr
    g0[2] = (unsigned)rfl((unsigned int)ga);                      // global_addr[31:0]
    g0[3] = (unsigned)rfl(((unsigned int)((ga >> 32) & 0x01FFFFFFull)) | (2u << 30)); // [56:32] | type=2
    v8s g1;
    g1[0] = rfl(0x00010000u);                                     // data_size=1 (2 bytes)
    g1[1] = rfl(((unsigned)KDIM & 0xFFFFu) << 16);                // tensor_dim0 lo16
    g1[2] = rfl((((unsigned)KDIM >> 16) & 0xFFFFu) | (((unsigned)NDIM & 0xFFFFu) << 16));
    g1[3] = rfl((((unsigned)NDIM >> 16) & 0xFFFFu) | (32u << 16)); // tile_dim0=32
    g1[4] = rfl(128u);                                            // tile_dim1=128, tile_dim2=0
    g1[5] = rfl((unsigned)KDIM);                                  // tensor_dim0_stride lo32
    g1[6] = rfl(0u);
    g1[7] = rfl(0u);
    v4s z = {0, 0, 0, 0};
#if __clang_major__ >= 23
    v8s z8 = {0, 0, 0, 0, 0, 0, 0, 0};
    __builtin_amdgcn_tensor_load_to_lds(g0, g1, z, z, z8, 0);
#else
    __builtin_amdgcn_tensor_load_to_lds(g0, g1, z, z, 0);
#endif
}
#endif

// ---------------------------------------------------------------------------
// Kernel 3: WMMA GEMM  out(4096 x 768, f32) = A(4096 x 9984, f16) * Wp^T
//   128x128 block tile, BK=32, double-buffered LDS.
//   A tiles: per-lane global_load_async_to_lds_b128 (ASYNCcnt).
//   B tiles: one TDM tensor_load_to_lds per stage from wave 0 (TENSORcnt).
//   8 waves x (2x4) v_wmma_f32_16x16x32_f16.
// ---------------------------------------------------------------------------
__global__ __launch_bounds__(256)
void gemm_kernel(const _Float16* __restrict__ Aw,
                 const _Float16* __restrict__ Wp,
                 float* __restrict__ out) {
    __shared__ __align__(16) _Float16 sA[2][128 * 32];
    __shared__ __align__(16) _Float16 sB[2][128 * 32];

    int t    = threadIdx.x;
    int lane = t & 31;
    int w    = t >> 5;
    int g    = lane >> 4;     // half-wave group (K split in fragments)
    int lr   = lane & 15;

    int bm = blockIdx.x;      // 0..31
    int bn = blockIdx.y;      // 0..5
    int wm = (w & 3) * 32;
    int wn = (w >> 2) * 64;

    const _Float16* gA = Aw + (size_t)(bm * 128) * KDIM;
    const _Float16* gB = Wp + (size_t)(bn * 128) * KDIM;

    // cooperative A-tile mapping: 512 x 16B chunks, 2 per thread
    int r0 = t >> 2, c0 = t & 3;
    int r1 = r0 + 64;         // (t+256)>>2 ; (t+256)&3 == c0

    const v8f vzero = {0.f, 0.f, 0.f, 0.f, 0.f, 0.f, 0.f, 0.f};
    v8f acc[2][4];
    #pragma unroll
    for (int i = 0; i < 2; ++i)
        #pragma unroll
        for (int j = 0; j < 4; ++j) acc[i][j] = vzero;

    auto issue_stage = [&](int kb, int buf) {
        async_copy_b128(gA + (size_t)r0 * KDIM + kb + c0 * 8, &sA[buf][r0 * 32 + c0 * 8]);
        async_copy_b128(gA + (size_t)r1 * KDIM + kb + c0 * 8, &sA[buf][r1 * 32 + c0 * 8]);
#if USE_TDM
        if (w == 0) tdm_load_tile(gB + kb, &sB[buf][0]);
#else
        async_copy_b128(gB + (size_t)r0 * KDIM + kb + c0 * 8, &sB[buf][r0 * 32 + c0 * 8]);
        async_copy_b128(gB + (size_t)r1 * KDIM + kb + c0 * 8, &sB[buf][r1 * 32 + c0 * 8]);
#endif
    };

    issue_stage(0, 0);

    const int NT = KDIM / 32;          // 312 stages
    for (int kt = 0; kt < NT; ++kt) {
        int cur = kt & 1;
        if (kt + 1 < NT) {
            issue_stage((kt + 1) * 32, cur ^ 1);
            // keep one stage in flight; wait for the current stage only
#if USE_TDM
            asm volatile("s_wait_asynccnt 2" ::: "memory");
            if (w == 0) wait_tensor1();
#else
            asm volatile("s_wait_asynccnt 4" ::: "memory");
#endif
        } else {
#if USE_TDM
            asm volatile("s_wait_asynccnt 0" ::: "memory");
            if (w == 0) wait_tensor0();
#else
            asm volatile("s_wait_asynccnt 0" ::: "memory");
#endif
        }
        __syncthreads();               // LDS stage `cur` visible to all waves

        const _Float16* cA = &sA[cur][0];
        const _Float16* cB = &sB[cur][0];
        FragH fa[2], fb[4];
        #pragma unroll
        for (int tm = 0; tm < 2; ++tm) {
            int row = wm + tm * 16 + lr;   // A frag: lanes 0-15 K0..7/16..23, lanes 16-31 K8..15/24..31
            fa[tm].q[0] = *(const uint4*)&cA[row * 32 + g * 8];
            fa[tm].q[1] = *(const uint4*)&cA[row * 32 + 16 + g * 8];
        }
        #pragma unroll
        for (int tn = 0; tn < 4; ++tn) {
            int col = wn + tn * 16 + lr;   // B frag: lanes 0-15 K0..7/16..23, lanes 16-31 K8..15/24..31
            fb[tn].q[0] = *(const uint4*)&cB[col * 32 + g * 8];
            fb[tn].q[1] = *(const uint4*)&cB[col * 32 + 16 + g * 8];
        }
        #pragma unroll
        for (int tm = 0; tm < 2; ++tm)
            #pragma unroll
            for (int tn = 0; tn < 4; ++tn)
                acc[tm][tn] = __builtin_amdgcn_wmma_f32_16x16x32_f16(
                    false, fa[tm].h, false, fb[tn].h,
                    (short)0, acc[tm][tn], false, false);

        __syncthreads();               // all reads of buffer `cur` done before refill
    }

    // D layout: element i -> M = i + 8*g, N = lr ; scatter to out[b,d,p,r]
    #pragma unroll
    for (int tm = 0; tm < 2; ++tm) {
        int mbase = bm * 128 + wm + tm * 16 + g * 8;
        #pragma unroll
        for (int tn = 0; tn < 4; ++tn) {
            int nd = bn * 128 + wn + tn * 16 + lr;
            int d  = nd / A_;
            int r  = nd - d * A_;
            #pragma unroll
            for (int i = 0; i < 8; ++i) {
                int m = mbase + i;
                int b = m >> 10;
                int p = m & (P_ - 1);
                out[(((size_t)b * C2_ + d) * P_ + p) * A_ + r] = acc[tm][tn][i];
            }
        }
    }
}

// ---------------------------------------------------------------------------
extern "C" void kernel_launch(void* const* d_in, const int* in_sizes, int n_in,
                              void* d_out, int out_size, void* d_ws, size_t ws_size,
                              hipStream_t stream) {
    const int*   nbr     = (const int*)  d_in[0];
    const float* verts   = (const float*)d_in[1];
    const float* fm      = (const float*)d_in[2];
    const float* W       = (const float*)d_in[3];
    const float* vs      = (const float*)d_in[4];
    const int*   idx_map = (const int*)  d_in[5];
    const int*   tr_kv   = (const int*)  d_in[6];
    const int*   tr_a    = (const int*)  d_in[7];
    float* out = (float*)d_out;

    _Float16* Aw = (_Float16*)d_ws;                                        // 4096*9984*2 B
    _Float16* Wp = (_Float16*)((char*)d_ws + (size_t)MDIM * KDIM * 2);     // 768*9984*2 B

    int totalW = NDIM * KDIM;
    build_w_kernel<<<(totalW + 255) / 256, 256, 0, stream>>>(W, idx_map, tr_kv, tr_a, Wp);
    build_a_kernel<<<MDIM, 256, 0, stream>>>(nbr, verts, fm, vs, Aw);
    dim3 grid(MDIM / 128, NDIM / 128);
    gemm_kernel<<<grid, 256, 0, stream>>>(Aw, Wp, out);
}
